// NAPPGATBase_75350906241478
// MI455X (gfx1250) — matrix operand order repaired and checked by reading, not compiled, not run on hardware
//
#include <hip/hip_runtime.h>
#include <hip/hip_bf16.h>

// ---------------------------------------------------------------------------
// NAPPGAT on gfx1250: GAT x3 + SAG gate + dense-pool losses.
// GEMMs: v_wmma_f32_16x16x32_bf16, operands pre-swizzled to fragment order
// (b128 global loads for A, LDS-staged b128 loads for W). NT is a template
// parameter so the 4x7 wmma chain is branch-free; nt-outer/kt-inner keeps a
// single live accumulator (no scratch spills).
// Edge segment ops: wave32-per-edge coalesced gather/scatter (HBM-bound).
// ---------------------------------------------------------------------------

#define HD   100      // hidden / embedding dim
#define KP   128      // padded K for WMMA (4 k-steps of 32)
#define NPGc 2000
#define CKc  16

typedef __bf16 v16bf __attribute__((ext_vector_type(16)));
typedef float  v8f   __attribute__((ext_vector_type(8)));

__device__ __forceinline__ unsigned short f2bf(float f) {
  unsigned u = __float_as_uint(f);
  u += 0x7FFFu + ((u >> 16) & 1u);          // round-to-nearest-even
  return (unsigned short)(u >> 16);
}

__device__ __forceinline__ void atomicMaxF32(float* addr, float val) {
  int iv = __float_as_int(val);
  if (iv >= 0) atomicMax((int*)addr, iv);
  else         atomicMin((unsigned int*)addr, (unsigned int)iv);
}

// ---------------- fill ----------------
__global__ void k_fill(float* p, float v, size_t n) {
  size_t i = (size_t)blockIdx.x * blockDim.x + threadIdx.x;
  size_t st = (size_t)gridDim.x * blockDim.x;
  for (; i < n; i += st) p[i] = v;
}

// -------- A swizzle: f32 (N x HD) -> bf16 fragment order ------------------
// Element o: j=o&15 (pos in lane), L=(o>>4)&31 (lane), kt=(o>>9)&3, t=o>>11.
// row = t*16 + (L&15); k = kt*32 + (j>>3)*16 + (L>>4)*8 + (j&7).
__global__ void k_swz_a(const float* __restrict__ src, unsigned short* __restrict__ dst,
                        int N) {
  size_t total = (size_t)N * KP;
  size_t i = (size_t)blockIdx.x * blockDim.x + threadIdx.x;
  size_t st = (size_t)gridDim.x * blockDim.x;
  for (; i < total; i += st) {
    int j = (int)(i & 15), L = (int)((i >> 4) & 31);
    int kt = (int)((i >> 9) & 3);
    size_t t = i >> 11;
    int row = (int)(t * 16) + (L & 15);
    int k = kt * 32 + ((j >> 3) << 4) + ((L >> 4) << 3) + (j & 7);
    float v = (k < HD) ? src[(size_t)row * HD + k] : 0.f;
    dst[i] = f2bf(v);
  }
}

// -------- embedding gather straight into A fragment order -----------------
__global__ void k_swz_embed(const int* __restrict__ x, const float* __restrict__ emb,
                            unsigned short* __restrict__ dst, int N) {
  size_t total = (size_t)N * KP;
  size_t i = (size_t)blockIdx.x * blockDim.x + threadIdx.x;
  size_t st = (size_t)gridDim.x * blockDim.x;
  for (; i < total; i += st) {
    int j = (int)(i & 15), L = (int)((i >> 4) & 31);
    int kt = (int)((i >> 9) & 3);
    size_t t = i >> 11;
    int row = (int)(t * 16) + (L & 15);
    int k = kt * 32 + ((j >> 3) << 4) + ((L >> 4) << 3) + (j & 7);
    float v = (k < HD) ? emb[(size_t)x[row * 3] * HD + k] : 0.f;
    dst[i] = f2bf(v);
  }
}

// -------- W swizzle: f32 (HD x C) -> bf16 B-fragment order ----------------
// Tile order (nt*4 + kt) so the kt-inner GEMM loop reads contiguous LDS.
// Element o: j=o&15, L=(o>>4)&31, tile=o>>9; nt=tile>>2, kt=tile&3.
// n = nt*16 + (L&15); k = kt*32 + (j>>3)*16 + (L>>4)*8 + (j&7).
__global__ void k_swz_w(const float* __restrict__ src, unsigned short* __restrict__ dst,
                        int NT, int C) {
  int total = 4 * NT * 512;
  int i = blockIdx.x * blockDim.x + threadIdx.x;
  int st = gridDim.x * blockDim.x;
  for (; i < total; i += st) {
    int j = i & 15, L = (i >> 4) & 31, tile = i >> 9;
    int nt = tile >> 2, kt = tile & 3;
    int n = nt * 16 + (L & 15);
    int k = kt * 32 + ((j >> 3) << 4) + ((L >> 4) << 3) + (j & 7);
    float v = (k < HD && n < C) ? src[(size_t)k * C + n] : 0.f;
    dst[i] = f2bf(v);
  }
}

// ---------------- WMMA GEMM: C[N x NC] = A[N x KP] * W[KP x NT*16] --------
// 8 waves/block, 16 rows/wave. A: 2x b128 global per frag; W staged in LDS.
// nt outer / kt inner: one live accumulator, store per finished column tile.
template <int NT>
__global__ void k_gemm_wmma(const uint4* __restrict__ Asw,
                            const uint4* __restrict__ Wsw,
                            float* __restrict__ C,
                            int nRows, int NC) {
  __shared__ uint4 lw[4 * NT * 64];          // NT=7 -> 28KB
  int tid = threadIdx.x;
#pragma unroll 1
  for (int i = tid; i < 4 * NT * 64; i += 256) lw[i] = Wsw[i];
  __syncthreads();

  int wave = tid >> 5, lane = tid & 31;
  int t = blockIdx.x * 8 + wave;
  if (t * 16 >= nRows) return;               // wave-uniform

  union frag { v16bf v; uint4 q[2]; };
  frag a[4];
  const uint4* ab = Asw + ((size_t)t * 4 * 32 + lane) * 2;
#pragma unroll
  for (int kt = 0; kt < 4; ++kt) {
    a[kt].q[0] = ab[(size_t)kt * 64];
    a[kt].q[1] = ab[(size_t)kt * 64 + 1];
  }

  int half = lane >> 4;
  int colLo = lane & 15;

#pragma unroll
  for (int nt = 0; nt < NT; ++nt) {
    v8f acc = (v8f){0.f, 0.f, 0.f, 0.f, 0.f, 0.f, 0.f, 0.f};
#pragma unroll
    for (int kt = 0; kt < 4; ++kt) {
      frag b;
      int idx = ((nt * 4 + kt) * 32 + lane) * 2;
      b.q[0] = lw[idx];
      b.q[1] = lw[idx + 1];
      acc = __builtin_amdgcn_wmma_f32_16x16x32_bf16(
          false, a[kt].v, false, b.v, (short)0, acc, false, false);
    }
    int col = nt * 16 + colLo;
    if (col < NC) {
#pragma unroll
      for (int r = 0; r < 8; ++r) {
        int mm = r + 8 * half;
        C[(size_t)(t * 16 + mm) * NC + col] = acc[r];
      }
    }
  }
}

// ---------------- attention logits (wave per node, shfl reduce) -----------
__global__ void k_al(const float* __restrict__ hp, const float* __restrict__ a_s,
                     const float* __restrict__ a_d,
                     float* al_s, float* al_d, int N) {
  int lane = threadIdx.x & 31;
  int wid = (blockIdx.x * blockDim.x + threadIdx.x) >> 5;
  int nw = (gridDim.x * blockDim.x) >> 5;
  for (int n = wid; n < N; n += nw) {
    const float* row = hp + (size_t)n * HD;
    float s = 0.f, d = 0.f;
    for (int f = lane; f < HD; f += 32) {
      float v = row[f];
      s += v * a_s[f];
      d += v * a_d[f];
    }
    for (int o = 16; o; o >>= 1) {
      s += __shfl_xor(s, o, 32);
      d += __shfl_xor(d, o, 32);
    }
    if (lane == 0) { al_s[n] = s; al_d[n] = d; }
  }
}

// ---------------- edge phase 1: leaky-relu score + segment max ----------
__global__ void k_edge_score(const int* src, const int* dst,
                             const float* al_s, const float* al_d,
                             float* ev, float* mx, int E, int ET) {
  int e = blockIdx.x * blockDim.x + threadIdx.x;
  int st = gridDim.x * blockDim.x;
  for (; e < ET; e += st) {
    int s = (e < E) ? src[e] : (e - E);
    int d = (e < E) ? dst[e] : (e - E);
    float v = al_s[s] + al_d[d];
    v = (v > 0.f) ? v : 0.2f * v;
    ev[e] = v;
    atomicMaxF32(&mx[d], v);
  }
}

// ---------------- edge phase 2: exp + segment sum ----------------
__global__ void k_edge_exp(const int* dst, float* ev, const float* mx,
                           float* den, int E, int ET) {
  int e = blockIdx.x * blockDim.x + threadIdx.x;
  int st = gridDim.x * blockDim.x;
  for (; e < ET; e += st) {
    int d = (e < E) ? dst[e] : (e - E);
    float x = __expf(ev[e] - mx[d]);
    ev[e] = x;
    atomicAdd(&den[d], x);
  }
}

// ------------- edge phase 3: alpha-weighted scatter (wave per edge) -------
__global__ void k_edge_aggr(const int* __restrict__ src, const int* __restrict__ dst,
                            const float* __restrict__ ev, const float* __restrict__ den,
                            const float* __restrict__ hp, float* out, int E, int ET) {
  int lane = threadIdx.x & 31;
  int wid = (blockIdx.x * blockDim.x + threadIdx.x) >> 5;
  int nw = (gridDim.x * blockDim.x) >> 5;
  for (int e = wid; e < ET; e += nw) {
    int s = (e < E) ? src[e] : (e - E);
    int d = (e < E) ? dst[e] : (e - E);
    float alpha = ev[e] / den[d];
    const float* hs = hp + (size_t)s * HD;
    float* od = out + (size_t)d * HD;
    for (int f = lane; f < HD; f += 32)
      atomicAdd(&od[f], alpha * hs[f]);
  }
}

// ---------------- out = relu(out + bias) ----------------
__global__ void k_bias_relu(float* out, const float* b, size_t total) {
  size_t i = (size_t)blockIdx.x * blockDim.x + threadIdx.x;
  size_t st = (size_t)gridDim.x * blockDim.x;
  for (; i < total; i += st) {
    float v = out[i] + b[i % HD];
    out[i] = v > 0.f ? v : 0.f;
  }
}

// ---------------- SAG neigh scatter (wave per edge, real edges) -----------
__global__ void k_scatter_neigh(const int* __restrict__ src, const int* __restrict__ dst,
                                const float* __restrict__ h, float* neigh, int E) {
  int lane = threadIdx.x & 31;
  int wid = (blockIdx.x * blockDim.x + threadIdx.x) >> 5;
  int nw = (gridDim.x * blockDim.x) >> 5;
  for (int e = wid; e < E; e += nw) {
    const float* hs = h + (size_t)src[e] * HD;
    float* nd = neigh + (size_t)dst[e] * HD;
    for (int f = lane; f < HD; f += 32)
      atomicAdd(&nd[f], hs[f]);
  }
}

// ------------- SAG gate: h *= tanh(neigh@Wrel + h@Wroot + b) --------------
__global__ void k_sag(const float* __restrict__ neigh, float* h,
                      const float* __restrict__ Wrel, const float* __restrict__ Wroot,
                      const float* __restrict__ bsag, int N) {
  int lane = threadIdx.x & 31;
  int wid = (blockIdx.x * blockDim.x + threadIdx.x) >> 5;
  int nw = (gridDim.x * blockDim.x) >> 5;
  for (int n = wid; n < N; n += nw) {
    const float* nr = neigh + (size_t)n * HD;
    float* hr = h + (size_t)n * HD;
    float s = 0.f;
    for (int f = lane; f < HD; f += 32) s += nr[f] * Wrel[f] + hr[f] * Wroot[f];
    for (int o = 16; o; o >>= 1) s += __shfl_xor(s, o, 32);
    float t = tanhf(s + bsag[0]);
    for (int f = lane; f < HD; f += 32) hr[f] *= t;
  }
}

// ---------------- softmax over CK=16 cluster logits ----------------
__global__ void k_softmax16(const float* slog, const float* bs, float* s, int N) {
  int n = blockIdx.x * blockDim.x + threadIdx.x;
  int st = gridDim.x * blockDim.x;
  for (; n < N; n += st) {
    float v[CKc], mx = -1e30f;
    for (int c = 0; c < CKc; ++c) { v[c] = slog[(size_t)n * CKc + c] + bs[c]; mx = fmaxf(mx, v[c]); }
    float sum = 0.f;
    for (int c = 0; c < CKc; ++c) { v[c] = __expf(v[c] - mx); sum += v[c]; }
    float inv = 1.f / sum;
    for (int c = 0; c < CKc; ++c) s[(size_t)n * CKc + c] = v[c] * inv;
  }
}

// ---------------- out-degree ----------------
__global__ void k_deg(const int* src, float* deg, int E) {
  int e = blockIdx.x * blockDim.x + threadIdx.x;
  int st = gridDim.x * blockDim.x;
  for (; e < E; e += st) atomicAdd(&deg[src[e]], 1.f);
}

// ---------------- out_adj[b] += outer(s[src], s[dst]) per edge ----------
// 256 edges per block; edges are graph-contiguous and EPG % 256 == 0.
__global__ void k_outadj(const int* src, const int* dst, const float* s,
                         float* oadj, int E, int EPG) {
  __shared__ float acc[CKc * CKc];
  int tid = threadIdx.x;
  acc[tid] = 0.f;
  __syncthreads();
  int e = blockIdx.x * 256 + tid;
  if (e < E) {
    const float* ss = s + (size_t)src[e] * CKc;
    const float* sd = s + (size_t)dst[e] * CKc;
    float sv[CKc], dv[CKc];
    for (int c = 0; c < CKc; ++c) { sv[c] = ss[c]; dv[c] = sd[c]; }
    for (int i = 0; i < CKc; ++i)
      for (int j = 0; j < CKc; ++j)
        atomicAdd(&acc[i * CKc + j], sv[i] * dv[j]);
  }
  __syncthreads();
  int b = (blockIdx.x * 256) / EPG;
  atomicAdd(&oadj[(size_t)b * 256 + tid], acc[tid]);
}

// ---------------- per-node pooled stats: ca, cluster_size, ss ----------
__global__ void k_node_stats(const float* s, const float* deg,
                             float* ca, float* cs, float* ssb, int N) {
  int n = blockIdx.x * blockDim.x + threadIdx.x;
  int st = gridDim.x * blockDim.x;
  for (; n < N; n += st) {
    int b = n / NPGc;
    float sv[CKc];
    for (int c = 0; c < CKc; ++c) sv[c] = s[(size_t)n * CKc + c];
    float dg = deg[n];
    for (int c = 0; c < CKc; ++c) {
      atomicAdd(&ca[b * CKc + c], sv[c] * dg);
      atomicAdd(&cs[b * CKc + c], sv[c]);
    }
    for (int i = 0; i < CKc; ++i)
      for (int j = 0; j < CKc; ++j)
        atomicAdd(&ssb[(size_t)b * 256 + i * CKc + j], sv[i] * sv[j]);
  }
}

// ---------------- final scalar loss (one wave, lane = graph) ----------
__global__ void k_final(const float* oadj, const float* ssb, const float* ca,
                        const float* cs, float* out, int B, float m2) {
  __shared__ float r0[32], r1[32], r2[32];
  int b = threadIdx.x;
  float spec = 0.f, orth = 0.f, clus = 0.f;
  if (b < B) {
    const float* oa = oadj + (size_t)b * 256;
    const float* sb = ssb + (size_t)b * 256;
    float tr = 0.f;
    for (int c = 0; c < CKc; ++c) tr += oa[c * CKc + c];
    float cc = 0.f;
    for (int c = 0; c < CKc; ++c) { float v = ca[b * CKc + c]; cc += v * v; }
    tr -= cc / m2;
    spec = -tr / m2;
    float fro = 0.f;
    for (int k = 0; k < 256; ++k) fro += sb[k] * sb[k];
    fro = sqrtf(fro);
    float o = 0.f;
    for (int i = 0; i < CKc; ++i)
      for (int j = 0; j < CKc; ++j) {
        float v = sb[i * CKc + j] / fro - (i == j ? 0.25f : 0.f);
        o += v * v;
      }
    orth = sqrtf(o);
    float c2 = 0.f;
    for (int c = 0; c < CKc; ++c) { float v = cs[b * CKc + c]; c2 += v * v; }
    clus = sqrtf(c2) / (float)NPGc * 4.f - 1.f;
  }
  r0[threadIdx.x] = spec; r1[threadIdx.x] = orth; r2[threadIdx.x] = clus;
  __syncthreads();
  if (threadIdx.x == 0) {
    float a = 0.f, bb = 0.f, c = 0.f;
    for (int i = 0; i < 32; ++i) { a += r0[i]; bb += r1[i]; c += r2[i]; }
    out[0] = (a + bb + c) / (float)B;
  }
}

// ===========================================================================
extern "C" void kernel_launch(void* const* d_in, const int* in_sizes, int n_in,
                              void* d_out, int out_size, void* d_ws, size_t ws_size,
                              hipStream_t stream) {
  const int*   x    = (const int*)d_in[0];
  const int*   ei   = (const int*)d_in[1];
  const float* emb  = (const float*)d_in[3];
  const float* W0   = (const float*)d_in[4];
  const float* as0  = (const float*)d_in[5];
  const float* ad0  = (const float*)d_in[6];
  const float* b0   = (const float*)d_in[7];
  const float* W1   = (const float*)d_in[8];
  const float* as1  = (const float*)d_in[9];
  const float* ad1  = (const float*)d_in[10];
  const float* b1   = (const float*)d_in[11];
  const float* Wrel = (const float*)d_in[12];
  const float* Wroot= (const float*)d_in[13];
  const float* bsag = (const float*)d_in[14];
  const float* W2   = (const float*)d_in[15];
  const float* as2  = (const float*)d_in[16];
  const float* ad2  = (const float*)d_in[17];
  const float* b2   = (const float*)d_in[18];
  const float* Ws   = (const float*)d_in[19];
  const float* bs   = (const float*)d_in[20];

  const int N  = in_sizes[2];           // 64000
  const int E  = in_sizes[1] / 2;       // 1,024,000
  const int ET = E + N;                 // + self loops
  const int Bg = N / NPGc;              // 32
  const int EPG = E / Bg;               // 32000
  const int* src = ei;
  const int* dst = ei + E;

  // ---- workspace layout ----
  char* p = (char*)d_ws;
  auto alloc = [&](size_t bytes) { char* q = p; p += (bytes + 255) & ~(size_t)255; return q; };
  unsigned short* Abf = (unsigned short*)alloc((size_t)N * KP * 2);
  unsigned short* Wbf = (unsigned short*)alloc((size_t)4 * 7 * 512 * 2);
  float* hp   = (float*)alloc((size_t)N * HD * 4);
  float* hA   = (float*)alloc((size_t)N * HD * 4);
  float* hB   = (float*)alloc((size_t)N * HD * 4);
  float* al_s = (float*)alloc((size_t)N * 4);
  float* al_d = (float*)alloc((size_t)N * 4);
  float* mx   = (float*)alloc((size_t)N * 4);
  float* den  = (float*)alloc((size_t)N * 4);
  float* ev   = (float*)alloc((size_t)ET * 4);
  float* sC   = (float*)alloc((size_t)N * CKc * 4);
  float* deg  = (float*)alloc((size_t)N * 4);
  float* oadj = (float*)alloc((size_t)Bg * 256 * 4);
  float* ssb  = (float*)alloc((size_t)Bg * 256 * 4);
  float* ca   = (float*)alloc((size_t)Bg * CKc * 4);
  float* cs   = (float*)alloc((size_t)Bg * CKc * 4);

  const int TB = 256;
  auto gb = [&](size_t total) { size_t g = (total + TB - 1) / TB; return (int)(g > 16384 ? 16384 : g); };
  const int gemmGrid = (N + 127) / 128;   // 8 waves/block, 16 rows/wave

  // Edge phases shared by all three GAT layers (hp already computed).
  auto gat_edges = [&](const float* asv, const float* adv, const float* bias,
                       float* h_out) {
    k_al<<<gb((size_t)N * 32), TB, 0, stream>>>(hp, asv, adv, al_s, al_d, N);
    k_fill<<<gb(N), TB, 0, stream>>>(mx, -INFINITY, (size_t)N);
    k_fill<<<gb(N), TB, 0, stream>>>(den, 0.f, (size_t)N);
    k_fill<<<gb((size_t)N * HD), TB, 0, stream>>>(h_out, 0.f, (size_t)N * HD);
    k_edge_score<<<gb(ET), TB, 0, stream>>>(src, dst, al_s, al_d, ev, mx, E, ET);
    k_edge_exp<<<gb(ET), TB, 0, stream>>>(dst, ev, mx, den, E, ET);
    k_edge_aggr<<<gb((size_t)ET * 32), TB, 0, stream>>>(src, dst, ev, den, hp, h_out, E, ET);
    k_bias_relu<<<gb((size_t)N * HD), TB, 0, stream>>>(h_out, bias, (size_t)N * HD);
  };

  // ---- layer 0: embed gather -> swizzled bf16, GEMM, edges -> hA ----
  k_swz_embed<<<gb((size_t)N * KP), TB, 0, stream>>>(x, emb, Abf, N);
  k_swz_w<<<gb(4 * 7 * 512), TB, 0, stream>>>(W0, Wbf, 7, HD);
  k_gemm_wmma<7><<<gemmGrid, 256, 0, stream>>>((const uint4*)Abf, (const uint4*)Wbf, hp, N, HD);
  gat_edges(as0, ad0, b0, hA);

  // ---- layer 1: hA -> hB ----
  k_swz_a<<<gb((size_t)N * KP), TB, 0, stream>>>(hA, Abf, N);
  k_swz_w<<<gb(4 * 7 * 512), TB, 0, stream>>>(W1, Wbf, 7, HD);
  k_gemm_wmma<7><<<gemmGrid, 256, 0, stream>>>((const uint4*)Abf, (const uint4*)Wbf, hp, N, HD);
  gat_edges(as1, ad1, b1, hB);

  // ---- SAG gate: neigh = scatter(hB[src] -> dst); hB *= tanh(score) ----
  k_fill<<<gb((size_t)N * HD), TB, 0, stream>>>(hA, 0.f, (size_t)N * HD);
  k_scatter_neigh<<<gb((size_t)E * 32), TB, 0, stream>>>(src, dst, hB, hA, E);
  k_sag<<<gb((size_t)N * 32), TB, 0, stream>>>(hA, hB, Wrel, Wroot, bsag, N);

  // ---- layer 2: hB -> hA ----
  k_swz_a<<<gb((size_t)N * KP), TB, 0, stream>>>(hB, Abf, N);
  k_swz_w<<<gb(4 * 7 * 512), TB, 0, stream>>>(W2, Wbf, 7, HD);
  k_gemm_wmma<7><<<gemmGrid, 256, 0, stream>>>((const uint4*)Abf, (const uint4*)Wbf, hp, N, HD);
  gat_edges(as2, ad2, b2, hA);

  // ---- pooling: s = softmax(hA @ Ws + bs) via WMMA (NT=1, NC=16) ----
  k_swz_a<<<gb((size_t)N * KP), TB, 0, stream>>>(hA, Abf, N);
  k_swz_w<<<gb(4 * 1 * 512), TB, 0, stream>>>(Ws, Wbf, 1, CKc);
  k_gemm_wmma<1><<<gemmGrid, 256, 0, stream>>>((const uint4*)Abf, (const uint4*)Wbf, hp, N, CKc);
  k_softmax16<<<gb(N), TB, 0, stream>>>(hp, bs, sC, N);

  k_fill<<<gb(N), TB, 0, stream>>>(deg, 0.f, (size_t)N);
  k_fill<<<1, TB, 0, stream>>>(oadj, 0.f, (size_t)Bg * 256);
  k_fill<<<1, TB, 0, stream>>>(ssb, 0.f, (size_t)Bg * 256);
  k_fill<<<1, TB, 0, stream>>>(ca, 0.f, (size_t)Bg * CKc);
  k_fill<<<1, TB, 0, stream>>>(cs, 0.f, (size_t)Bg * CKc);

  k_deg<<<gb(E), TB, 0, stream>>>(src, deg, E);
  k_outadj<<<(E + 255) / 256, 256, 0, stream>>>(src, dst, sC, oadj, E, EPG);
  k_node_stats<<<gb(N), TB, 0, stream>>>(sC, deg, ca, cs, ssb, N);
  k_final<<<1, 32, 0, stream>>>(oadj, ssb, ca, cs, (float*)d_out, Bg, (float)EPG);
}
